// SparseMoE_10024453669471
// MI455X (gfx1250) — matrix operand order repaired and checked by reading, not compiled, not run on hardware
//
#include <hip/hip_runtime.h>
#include <hip/hip_bf16.h>

// Sparse MoE forward for MI455X (gfx1250, wave32).
// router (VALU) -> bf16 x -> grouped WMMA GEMMs (v_wmma_f32_16x16x32_bf16)
// h = silu(x@W1)*(x@W3) staged bf16 in ws, y += w * h@W2 via f32 atomics.
// v4: fix global_load_async_to_lds_b128 arg types (toolchain wants generic int4*
// for the global src; LDS dst cast to AS(3) int4*, which also converts implicitly
// if the param is generic). Async A-tile double buffer + s_wait_asynccnt kept.

#define T_TOK 4096
#define DDIM  2048
#define EEXP  8
#define FDIM  4096
#define KC    32
#define LDK   40   // padded LDS row length in bf16 elems (80B: bank-spread, 16B aligned)
#define ASZ   (64 * LDK)

typedef __attribute__((ext_vector_type(16))) __bf16 v16bf;
typedef __attribute__((ext_vector_type(8)))  __bf16 v8bf;
typedef __attribute__((ext_vector_type(2)))  __bf16 v2bf;
typedef __attribute__((ext_vector_type(8)))  float  v8f;
typedef __attribute__((ext_vector_type(4)))  int    v4i;

#if __has_builtin(__builtin_amdgcn_global_load_async_to_lds_b128) && \
    __has_builtin(__builtin_amdgcn_s_wait_asynccnt)
#define HAVE_ASYNC_LDS 1
#else
#define HAVE_ASYNC_LDS 0
#endif

#if HAVE_ASYNC_LDS
// 16B global -> LDS async copy (ASYNCcnt-tracked)
__device__ __forceinline__ void async_copy16(const void* gsrc, void* ldst) {
  __builtin_amdgcn_global_load_async_to_lds_b128(
      (v4i*)gsrc, (__attribute__((address_space(3))) v4i*)ldst, 0, 0);
}
#endif

__device__ __forceinline__ __bf16 bf16rne(float f) {
  unsigned int u = __builtin_bit_cast(unsigned int, f);
  u += 0x7fffu + ((u >> 16) & 1u);                 // round-to-nearest-even
  unsigned short s = (unsigned short)(u >> 16);
  return __builtin_bit_cast(__bf16, s);
}

// store two consecutive bf16 values (dst 4B-aligned)
__device__ __forceinline__ void store_bf16x2(__bf16* dst, float a, float b) {
#if __has_builtin(__builtin_amdgcn_cvt_pk_bf16_f32)
  auto pk = __builtin_amdgcn_cvt_pk_bf16_f32(a, b);   // v_cvt_pk_bf16_f32 (RNE)
  *(v2bf*)dst = __builtin_bit_cast(v2bf, pk);
#else
  v2bf p;
  p[0] = bf16rne(a);
  p[1] = bf16rne(b);
  *(v2bf*)dst = p;
#endif
}

__device__ __forceinline__ v16bf cat8(v8bf a, v8bf b) {
  return __builtin_shufflevector(a, b, 0,1,2,3,4,5,6,7,8,9,10,11,12,13,14,15);
}

// ---------------------------------------------------------------- zero / cvt
__global__ void zero_kernel(float* __restrict__ y, size_t n, int* __restrict__ counts) {
  size_t i = (size_t)blockIdx.x * blockDim.x + threadIdx.x;
  const size_t stride = (size_t)gridDim.x * blockDim.x;
  for (; i < n; i += stride) y[i] = 0.0f;
  if (blockIdx.x == 0 && threadIdx.x < EEXP) counts[threadIdx.x] = 0;
}

__global__ void cvtx_kernel(const float* __restrict__ x, unsigned short* __restrict__ xb, size_t n) {
  size_t i = (size_t)blockIdx.x * blockDim.x + threadIdx.x;
  const size_t stride = (size_t)gridDim.x * blockDim.x;
  for (; i < n; i += stride)
    xb[i] = __builtin_bit_cast(unsigned short, bf16rne(x[i]));
}

// ---------------------------------------------------------------- router
// one wave32 per token: logits[8], softmax, top-2, atomic append of row id 2t+k
__global__ __launch_bounds__(128)
void router_kernel(const float* __restrict__ x, const float* __restrict__ Wg,
                   float* __restrict__ logits, int* __restrict__ counts,
                   int* __restrict__ lists, float* __restrict__ wts) {
  const int lane = threadIdx.x & 31;
  const int t    = blockIdx.x * 4 + (threadIdx.x >> 5);

  float acc[EEXP];
#pragma unroll
  for (int e = 0; e < EEXP; ++e) acc[e] = 0.0f;

  for (int i = 0; i < DDIM; i += 32) {
    const float xv = x[(size_t)t * DDIM + i + lane];
#pragma unroll
    for (int e = 0; e < EEXP; ++e)
      acc[e] += xv * Wg[e * DDIM + i + lane];
  }
#pragma unroll
  for (int e = 0; e < EEXP; ++e)
#pragma unroll
    for (int off = 16; off > 0; off >>= 1)
      acc[e] += __shfl_xor(acc[e], off, 32);

  if (lane == 0) {
    float mx = acc[0];
#pragma unroll
    for (int e = 1; e < EEXP; ++e) mx = fmaxf(mx, acc[e]);
    float p[EEXP];
#pragma unroll
    for (int e = 0; e < EEXP; ++e) p[e] = __expf(acc[e] - mx);
    int i0 = 0;
#pragma unroll
    for (int e = 1; e < EEXP; ++e) if (p[e] > p[i0]) i0 = e;   // first max wins (JAX tie rule)
    int i1 = (i0 == 0) ? 1 : 0;
#pragma unroll
    for (int e = 0; e < EEXP; ++e) if (e != i0 && p[e] > p[i1]) i1 = e;
    const float inv = 1.0f / (p[i0] + p[i1]);                  // softmax denom cancels
#pragma unroll
    for (int e = 0; e < EEXP; ++e) logits[(size_t)t * EEXP + e] = acc[e];
    const int s0 = atomicAdd(&counts[i0], 1);
    lists[i0 * T_TOK + s0] = t * 2;
    const int s1 = atomicAdd(&counts[i1], 1);
    lists[i1 * T_TOK + s1] = t * 2 + 1;
    wts[2 * t]     = p[i0] * inv;
    wts[2 * t + 1] = p[i1] * inv;
  }
}

// ---------------------------------------------------------------- h = silu(x@W1)*(x@W3)
// block tile: 64 tokens x 128 F-cols, K chunks of 32 over D. 8 waves, each wave:
// 1 A-frag, 4x(W1,W3) 16x16 accumulators => 8 v_wmma per K step per wave.
__global__ __launch_bounds__(256)
void moe_h_kernel(const unsigned short* __restrict__ xb,
                  const float* __restrict__ W1, const float* __restrict__ W3,
                  const int* __restrict__ lists, const int* __restrict__ counts,
                  unsigned short* __restrict__ h) {
  __shared__ alignas(16) __bf16 As[2 * ASZ];     // double-buffered A tile
  __shared__ alignas(16) __bf16 B1s[128 * LDK];
  __shared__ alignas(16) __bf16 B3s[128 * LDK];
  __shared__ int rowIds[64];

  const int e     = blockIdx.x >> 6;
  const int tile0 = (blockIdx.x & 63) * 64;
  const int cnt   = counts[e];
  if (tile0 >= cnt) return;                     // block-uniform exit
  const int n0 = blockIdx.y * 128;

  const int tid  = threadIdx.x;
  const int lane = tid & 31;
  const int wave = tid >> 5;
  const int msub = wave & 3;                    // 4 M-subtiles of 16
  const int ngrp = wave >> 2;                   // 2 groups x 4 N-subtiles

  if (tid < 64) {
    int slot = tile0 + tid;
    if (slot >= cnt) slot = cnt - 1;            // clamp; stores are guarded below
    rowIds[tid] = lists[e * T_TOK + slot];      // row id = 2t+k
  }
  __syncthreads();                              // rowIds visible before first A load

  v8f acc1[4], acc3[4];
#pragma unroll
  for (int j = 0; j < 4; ++j) { acc1[j] = (v8f)(0.0f); acc3[j] = (v8f)(0.0f); }

  const size_t wbase = (size_t)e * DDIM * FDIM;
  const int fl = tid & 127;                     // F column within tile
  const int dg = (tid >> 7) * 16;               // half of the K chunk
  const int half = (lane < 16) ? 0 : 1;
  const int ar = tid >> 2;                      // A tile row this thread loads
  const int ac = (tid & 3) * 8;                 // A tile col this thread loads
  const unsigned short* aptr = xb + (size_t)(rowIds[ar] >> 1) * DDIM + ac;
  const float* p1 = W1 + wbase + (size_t)dg * FDIM + n0 + fl;
  const float* p3 = W3 + wbase + (size_t)dg * FDIM + n0 + fl;

  // ---- prologue: stage tile kt=0
  float w1Reg[16], w3Reg[16];
#pragma unroll
  for (int dd = 0; dd < 16; ++dd) {
    w1Reg[dd] = p1[(size_t)dd * FDIM];
    w3Reg[dd] = p3[(size_t)dd * FDIM];
  }
#if HAVE_ASYNC_LDS
  async_copy16(aptr, As + ar * LDK + ac);
#else
  uint4 aReg = *(const uint4*)aptr;
#endif

  for (int kt = 0; kt < DDIM / KC; ++kt) {
    const bool more = (kt + 1 < DDIM / KC);
    __syncthreads();                            // previous compute done
    // ---- commit staged weight registers to LDS (packed bf16 stores)
#if !HAVE_ASYNC_LDS
    *(uint4*)(As + ar * LDK + ac) = aReg;
#endif
#pragma unroll
    for (int dd = 0; dd < 16; dd += 2) {
      store_bf16x2(B1s + fl * LDK + dg + dd, w1Reg[dd], w1Reg[dd + 1]);
      store_bf16x2(B3s + fl * LDK + dg + dd, w3Reg[dd], w3Reg[dd + 1]);
    }

    // ---- issue next tile's global loads (latency hides behind the WMMAs)
    if (more) {
#if HAVE_ASYNC_LDS
      async_copy16(aptr + (kt + 1) * KC, As + ((kt + 1) & 1) * ASZ + ar * LDK + ac);
#else
      aReg = *(const uint4*)(aptr + (kt + 1) * KC);
#endif
      const float* q1 = p1 + (size_t)(kt + 1) * KC * FDIM;
      const float* q3 = p3 + (size_t)(kt + 1) * KC * FDIM;
#pragma unroll
      for (int dd = 0; dd < 16; ++dd) {
        w1Reg[dd] = q1[(size_t)dd * FDIM];
        w3Reg[dd] = q3[(size_t)dd * FDIM];
      }
      if (kt + 2 < DDIM / KC) {                 // global_prefetch_b8: warm L2 one more tile out
        __builtin_prefetch(q1 + (size_t)KC * FDIM, 0, 0);
        __builtin_prefetch(q3 + (size_t)KC * FDIM, 0, 0);
      }
    }
#if HAVE_ASYNC_LDS
    if (more) __builtin_amdgcn_s_wait_asynccnt(1);  // A(kt) landed, A(kt+1) in flight
    else      __builtin_amdgcn_s_wait_asynccnt(0);
#endif
    __syncthreads();

    // ---- fragments: batch the ds loads, then issue WMMAs back-to-back
    // A frag: lane<16 -> K 0-7 & 16-23 of row M=lane; lane>=16 -> K 8-15 & 24-31
#if HAVE_ASYNC_LDS
    const __bf16* arow = As + (kt & 1) * ASZ + (msub * 16 + (lane & 15)) * LDK;
#else
    const __bf16* arow = As + (msub * 16 + (lane & 15)) * LDK;
#endif
    const v16bf afrag = cat8(*(const v8bf*)(arow + half * 8),
                             *(const v8bf*)(arow + 16 + half * 8));
    v16bf bfrag[4];
#pragma unroll
    for (int j = 0; j < 4; ++j) {               // B frag: col N = lane&15, K halves by lane
      const __bf16* br = B1s + ((ngrp * 4 + j) * 16 + (lane & 15)) * LDK + half * 8;
      bfrag[j] = cat8(*(const v8bf*)br, *(const v8bf*)(br + 16));
    }
#pragma unroll
    for (int j = 0; j < 4; ++j)
      acc1[j] = __builtin_amdgcn_wmma_f32_16x16x32_bf16(false, afrag, false, bfrag[j],
                                                        (short)0, acc1[j], false, false);
#pragma unroll
    for (int j = 0; j < 4; ++j) {
      const __bf16* br = B3s + ((ngrp * 4 + j) * 16 + (lane & 15)) * LDK + half * 8;
      bfrag[j] = cat8(*(const v8bf*)br, *(const v8bf*)(br + 16));
    }
#pragma unroll
    for (int j = 0; j < 4; ++j)
      acc3[j] = __builtin_amdgcn_wmma_f32_16x16x32_bf16(false, afrag, false, bfrag[j],
                                                        (short)0, acc3[j], false, false);
  }

  // epilogue: h = silu(g)*u, bf16 store. C layout: VGPR i -> M = i (+8 for lane>=16), N = lane&15
#pragma unroll
  for (int j = 0; j < 4; ++j) {
    const int col = n0 + (ngrp * 4 + j) * 16 + (lane & 15);
#pragma unroll
    for (int i = 0; i < 8; ++i) {
      const int rl = msub * 16 + ((lane < 16) ? i : 8 + i);
      if (tile0 + rl < cnt) {
        const float g = acc1[j][i];
        const float u = acc3[j][i];
        const float s = g / (1.0f + __expf(-g));
        h[(size_t)rowIds[rl] * FDIM + col] =
            __builtin_bit_cast(unsigned short, bf16rne(s * u));
      }
    }
  }
}

// ---------------------------------------------------------------- y += w * (h @ W2)
__global__ __launch_bounds__(256)
void moe_y_kernel(const unsigned short* __restrict__ h,
                  const float* __restrict__ W2,
                  const int* __restrict__ lists, const int* __restrict__ counts,
                  const float* __restrict__ wts,
                  float* __restrict__ y) {
  __shared__ alignas(16) __bf16 As[2 * ASZ];
  __shared__ alignas(16) __bf16 Bs[128 * LDK];
  __shared__ int   rowIds[64];
  __shared__ float wrow[64];

  const int e     = blockIdx.x >> 6;
  const int tile0 = (blockIdx.x & 63) * 64;
  const int cnt   = counts[e];
  if (tile0 >= cnt) return;
  const int n0 = blockIdx.y * 128;

  const int tid  = threadIdx.x;
  const int lane = tid & 31;
  const int wave = tid >> 5;
  const int msub = wave & 3;
  const int ngrp = wave >> 2;

  if (tid < 64) {
    int slot = tile0 + tid;
    if (slot >= cnt) slot = cnt - 1;
    const int id = lists[e * T_TOK + slot];
    rowIds[tid] = id;
    wrow[tid]   = wts[id];
  }
  __syncthreads();

  v8f acc[4];
#pragma unroll
  for (int j = 0; j < 4; ++j) acc[j] = (v8f)(0.0f);

  const size_t wbase = (size_t)e * FDIM * DDIM;
  const int fl = tid & 127;
  const int dg = (tid >> 7) * 16;
  const int half = (lane < 16) ? 0 : 1;
  const int ar = tid >> 2;
  const int ac = (tid & 3) * 8;
  const unsigned short* aptr = h + (size_t)rowIds[ar] * FDIM + ac;
  const float* p2 = W2 + wbase + (size_t)dg * DDIM + n0 + fl;

  float w2Reg[16];
#pragma unroll
  for (int dd = 0; dd < 16; ++dd) w2Reg[dd] = p2[(size_t)dd * DDIM];
#if HAVE_ASYNC_LDS
  async_copy16(aptr, As + ar * LDK + ac);
#else
  uint4 aReg = *(const uint4*)aptr;
#endif

  for (int kt = 0; kt < FDIM / KC; ++kt) {
    const bool more = (kt + 1 < FDIM / KC);
    __syncthreads();
#if !HAVE_ASYNC_LDS
    *(uint4*)(As + ar * LDK + ac) = aReg;
#endif
#pragma unroll
    for (int dd = 0; dd < 16; dd += 2)
      store_bf16x2(Bs + fl * LDK + dg + dd, w2Reg[dd], w2Reg[dd + 1]);

    if (more) {
#if HAVE_ASYNC_LDS
      async_copy16(aptr + (kt + 1) * KC, As + ((kt + 1) & 1) * ASZ + ar * LDK + ac);
#else
      aReg = *(const uint4*)(aptr + (kt + 1) * KC);
#endif
      const float* q2 = p2 + (size_t)(kt + 1) * KC * DDIM;
#pragma unroll
      for (int dd = 0; dd < 16; ++dd) w2Reg[dd] = q2[(size_t)dd * DDIM];
      if (kt + 2 < FDIM / KC)
        __builtin_prefetch(q2 + (size_t)KC * DDIM, 0, 0);
    }
#if HAVE_ASYNC_LDS
    if (more) __builtin_amdgcn_s_wait_asynccnt(1);
    else      __builtin_amdgcn_s_wait_asynccnt(0);
#endif
    __syncthreads();

#if HAVE_ASYNC_LDS
    const __bf16* arow = As + (kt & 1) * ASZ + (msub * 16 + (lane & 15)) * LDK;
#else
    const __bf16* arow = As + (msub * 16 + (lane & 15)) * LDK;
#endif
    const v16bf afrag = cat8(*(const v8bf*)(arow + half * 8),
                             *(const v8bf*)(arow + 16 + half * 8));
    v16bf bfrag[4];
#pragma unroll
    for (int j = 0; j < 4; ++j) {
      const __bf16* br = Bs + ((ngrp * 4 + j) * 16 + (lane & 15)) * LDK + half * 8;
      bfrag[j] = cat8(*(const v8bf*)br, *(const v8bf*)(br + 16));
    }
#pragma unroll
    for (int j = 0; j < 4; ++j)
      acc[j] = __builtin_amdgcn_wmma_f32_16x16x32_bf16(false, afrag, false, bfrag[j],
                                                       (short)0, acc[j], false, false);
  }

  // scale by gate weight, combine into y (exactly 2 commutative f32 adds per element)
#pragma unroll
  for (int j = 0; j < 4; ++j) {
    const int col = n0 + (ngrp * 4 + j) * 16 + (lane & 15);
#pragma unroll
    for (int i = 0; i < 8; ++i) {
      const int rl = msub * 16 + ((lane < 16) ? i : 8 + i);
      if (tile0 + rl < cnt) {
        const int tok = rowIds[rl] >> 1;
        atomicAdd(&y[(size_t)tok * DDIM + col], acc[j][i] * wrow[rl]);
      }
    }
  }
}

// ---------------------------------------------------------------- launch
extern "C" void kernel_launch(void* const* d_in, const int* in_sizes, int n_in,
                              void* d_out, int out_size, void* d_ws, size_t ws_size,
                              hipStream_t stream) {
  (void)in_sizes; (void)n_in; (void)out_size; (void)ws_size;
  const float* x  = (const float*)d_in[0];
  const float* Wg = (const float*)d_in[1];
  const float* W1 = (const float*)d_in[2];
  const float* W3 = (const float*)d_in[3];
  const float* W2 = (const float*)d_in[4];

  float* y      = (float*)d_out;
  float* logits = y + (size_t)T_TOK * DDIM;

  // ws layout (~81 MB): counts | lists | gate wts | x_bf16 | h_bf16
  char* ws = (char*)d_ws;
  int*   counts = (int*)ws;                                          // 32 B (pad 256)
  int*   lists  = (int*)(ws + 256);                                  // E*T*4   = 128 KB
  float* wts    = (float*)(ws + 256 + EEXP * T_TOK * 4);             // 2T*4    = 32 KB
  unsigned short* xb = (unsigned short*)(ws + 256 + EEXP * T_TOK * 4 + 2 * T_TOK * 4); // T*D*2 = 16 MB
  unsigned short* hb = xb + (size_t)T_TOK * DDIM;                    // 2T*F*2  = 64 MB

  zero_kernel<<<2048, 256, 0, stream>>>(y, (size_t)T_TOK * DDIM, counts);
  cvtx_kernel<<<2048, 256, 0, stream>>>(x, xb, (size_t)T_TOK * DDIM);
  router_kernel<<<T_TOK / 4, 128, 0, stream>>>(x, Wg, logits, counts, lists, wts);
  moe_h_kernel<<<dim3(EEXP * (T_TOK / 64), FDIM / 128), 256, 0, stream>>>(
      xb, W1, W3, lists, counts, hb);
  moe_y_kernel<<<dim3(EEXP * (T_TOK / 64), DDIM / 128), 256, 0, stream>>>(
      hb, W2, lists, counts, wts, y);
}
